// Skinning_22290880266410
// MI455X (gfx1250) — matrix operand order repaired and checked by reading, not compiled, not run on hardware
//
#include <hip/hip_runtime.h>

// Fused LBS skinning as a single f16 WMMA GEMM:
//   out[v, (b,i)] = sum_{j,c} (W[v,j] * vh[v,c]) * M[b,j,i,c]
//   A : V x 208   (built on the fly, f16)
//   B : 208 x 96  (transform reshuffle, staged pre-swizzled in LDS, f16)
//   D : V x 96    (f32 accumulate), staged through LDS, written as
//                 coalesced b128 stores to the [B,V,3] layout.

typedef __attribute__((ext_vector_type(16))) _Float16 v16h;
typedef __attribute__((ext_vector_type(8)))  float    v8f;
typedef __attribute__((ext_vector_type(4)))  float    v4f;

#define JNUM 52
#define BNUM 32
#define NT_COUNT 6          // N = BNUM*3 = 96 -> 6 tiles of 16
#define KB_COUNT 7          // K = J*4 = 208 padded to 224 = 7 * 32

#define SB_HALFS (KB_COUNT * NT_COUNT * 512)      // 21504 halfs = 43008 B
#define SW_OFF   (SB_HALFS * 2)                   // 43008 (16B aligned)
#define SW_FLOATS (128 * JNUM)                    // 6656 floats = 26624 B
#define SVH_OFF  (SW_OFF + SW_FLOATS * 4)         // 69632
#define SMEM_BYTES (SVH_OFF + 128 * 4 * 4)        // 71680 B total LDS

#define OUT_PITCH 100                             // padded row pitch (floats) for D staging
// D staging buffer: 128 * 100 * 4 = 51200 B, overlays sB/sW after a barrier.

__global__ __launch_bounds__(256)
void lbs_wmma_kernel(const float* __restrict__ verts,   // [V,3]
                     const float* __restrict__ W,       // [V,J]
                     const float* __restrict__ M,       // [B,J,4,4]
                     float* __restrict__ out,           // [B,V,3]
                     int V)
{
    extern __shared__ char smem[];
    _Float16* sB  = (_Float16*)smem;              // WMMA-B fragment layout
    float*    sW  = (float*)(smem + SW_OFF);      // [128][52]
    float*    sVh = (float*)(smem + SVH_OFF);     // [128][4]
    float*    sOut = (float*)smem;                // phase 2: [128][OUT_PITCH]

    const int tid    = threadIdx.x;
    const int wave   = tid >> 5;
    const int lane   = tid & 31;
    const int vBlock = blockIdx.x * 128;

    // ---- stage B (transforms) pre-swizzled into 16x16x32 f16 B-fragment layout ----
    // chunk (kb,nt): 32 lanes x 16 halfs; lane ln, half e -> K = e + (ln>=16 ? 16 : 0),
    // N = nt*16 + (ln&15);  k = kb*32+K;  j = k>>2; c = k&3;  n -> (b = n/3, i = n%3)
    for (int idx = tid; idx < SB_HALFS; idx += 256) {
        int e  = idx & 15;
        int ln = (idx >> 4) & 31;
        int nt = (idx >> 9) % NT_COUNT;
        int kb = idx / (512 * NT_COUNT);
        int k  = kb * 32 + e + ((ln >= 16) ? 16 : 0);
        int j  = k >> 2;
        int c  = k & 3;
        int n  = nt * 16 + (ln & 15);
        int b  = n / 3;
        int i  = n - b * 3;
        float val = (j < JNUM) ? M[(b * JNUM + j) * 16 + i * 4 + c] : 0.0f;
        sB[idx] = (_Float16)val;
    }

    // ---- stage weights: coalesced copy of 128 contiguous rows ----
    {
        const long gbase = (long)vBlock * JNUM;
        const long gmax  = (long)V * JNUM;
        for (int idx = tid; idx < SW_FLOATS; idx += 256) {
            long g = gbase + idx;
            sW[idx] = (g < gmax) ? W[g] : 0.0f;
        }
    }

    // ---- stage homogeneous vertices (x,y,z,1) ----
    if (tid < 128) {
        int v = vBlock + tid;
        if (v >= V) v = V - 1;
        sVh[tid * 4 + 0] = verts[(long)v * 3 + 0];
        sVh[tid * 4 + 1] = verts[(long)v * 3 + 1];
        sVh[tid * 4 + 2] = verts[(long)v * 3 + 2];
        sVh[tid * 4 + 3] = 1.0f;
    }

    __syncthreads();

    // ---- per-wave 16(M) x 96(N) strip ----
    const int row    = wave * 16 + (lane & 15);    // A-matrix row for this lane
    const int laneHi = (lane >= 16) ? 8 : 0;       // M (D row) offset for upper half-wave
    const float vh0 = sVh[row * 4 + 0];
    const float vh1 = sVh[row * 4 + 1];
    const float vh2 = sVh[row * 4 + 2];
    const float* wrow = &sW[row * JNUM];

    v8f acc[NT_COUNT];
    #pragma unroll
    for (int nt = 0; nt < NT_COUNT; ++nt) acc[nt] = (v8f){};

    #pragma unroll
    for (int kb = 0; kb < KB_COUNT; ++kb) {
        // A fragment: half e -> K_local = e + (lane>=16 ? 16 : 0)
        // groups of 4 halves share one j (c = 0..3), vh[3] == 1
        v16h a;
        const int kbase = kb * 32 + ((lane >= 16) ? 16 : 0);
        #pragma unroll
        for (int g = 0; g < 4; ++g) {
            int Kl = g * 4;
            int j  = (kbase + Kl) >> 2;
            float w = (j < JNUM) ? wrow[j] : 0.0f;
            a[4 * g + 0] = (_Float16)(w * vh0);
            a[4 * g + 1] = (_Float16)(w * vh1);
            a[4 * g + 2] = (_Float16)(w * vh2);
            a[4 * g + 3] = (_Float16)w;
        }
        #pragma unroll
        for (int nt = 0; nt < NT_COUNT; ++nt) {
            v16h bfrag = *(const v16h*)(sB + (kb * NT_COUNT + nt) * 512 + lane * 16);
            acc[nt] = __builtin_amdgcn_wmma_f32_16x16x32_f16(
                false, a, false, bfrag, (short)0, acc[nt], false, false);
        }
    }

    // ---- stage D tile into LDS (overlays sB/sW; sVh region untouched) ----
    __syncthreads();   // all sB/sW reads complete before overwrite

    // D layout: VGPR r holds row M = r + laneHi, column N = lane&15
    {
        const int nlo = lane & 15;
        #pragma unroll
        for (int nt = 0; nt < NT_COUNT; ++nt) {
            int n = nt * 16 + nlo;
            #pragma unroll
            for (int r = 0; r < 8; ++r) {
                int vl = wave * 16 + laneHi + r;
                sOut[vl * OUT_PITCH + n] = acc[nt][r];
            }
        }
    }

    __syncthreads();

    // ---- coalesced write-out: per batch b, 384 contiguous floats = 96 float4 ----
    // global float index = b*V*3 + vBlock*3 + f,  f = vl*3 + i  (0..383)
    const bool fullTile = (vBlock + 128 <= V) && (((long)V * 3 & 3) == 0);
    if (fullTile) {
        for (int c = tid; c < BNUM * 96; c += 256) {
            int b     = c / 96;
            int chunk = c - b * 96;
            int f0    = chunk * 4;
            v4f val;
            {
                int vl0 = f0 / 3, i0 = f0 - vl0 * 3;
                int f1 = f0 + 1, vl1 = f1 / 3, i1 = f1 - vl1 * 3;
                int f2 = f0 + 2, vl2 = f2 / 3, i2 = f2 - vl2 * 3;
                int f3 = f0 + 3, vl3 = f3 / 3, i3 = f3 - vl3 * 3;
                val.x = sOut[vl0 * OUT_PITCH + b * 3 + i0];
                val.y = sOut[vl1 * OUT_PITCH + b * 3 + i1];
                val.z = sOut[vl2 * OUT_PITCH + b * 3 + i2];
                val.w = sOut[vl3 * OUT_PITCH + b * 3 + i3];
            }
            long g = (long)b * V * 3 + (long)vBlock * 3 + f0;
            __builtin_nontemporal_store(val, (v4f*)&out[g]);
        }
    } else {
        // partial last tile: guarded scalar stores
        for (int c = tid; c < BNUM * 96; c += 256) {
            int b     = c / 96;
            int chunk = c - b * 96;
            int f0    = chunk * 4;
            #pragma unroll
            for (int u = 0; u < 4; ++u) {
                int f  = f0 + u;
                int vl = f / 3;
                int i  = f - vl * 3;
                int v  = vBlock + vl;
                if (v < V) {
                    out[(long)b * V * 3 + (long)v * 3 + i] =
                        sOut[vl * OUT_PITCH + b * 3 + i];
                }
            }
        }
    }
}

extern "C" void kernel_launch(void* const* d_in, const int* in_sizes, int n_in,
                              void* d_out, int out_size, void* d_ws, size_t ws_size,
                              hipStream_t stream) {
    const float* verts = (const float*)d_in[0];   // vertices [1,V,3]
    const float* W     = (const float*)d_in[1];   // lbs_weights [1,V,J]
    const float* M     = (const float*)d_in[2];   // skinning_transforms [B,J,4,4]
    float* out = (float*)d_out;                   // [B,V,3]

    int V = in_sizes[0] / 3;
    int grid = (V + 127) / 128;
    lbs_wmma_kernel<<<grid, 256, SMEM_BYTES, stream>>>(verts, W, M, out, V);
}
